// TokenChoiceTopKRouter_1992864825974
// MI455X (gfx1250) — matrix operand order repaired and checked by reading, not compiled, not run on hardware
//
#include <hip/hip_runtime.h>
#include <hip/hip_bf16.h>

// ---------------------------------------------------------------------------
// MoE token-choice top-8 router for MI455X (gfx1250, wave32, WMMA).
//
//   K0 split_w_kernel : one-time gate_w fp32 -> bf16 hi/lo split (L2-resident)
//   K1 router_kernel  : bf16x3 split GEMM on v_wmma_f32_16x16x32_bf16
//                       (4 accumulators/wave, K-split x2, LDS reduction)
//                       + softmax + top-8 + per-block expert histogram
//   K2 prefix_kernel  : expert totals, exclusive scan, per-block sort offsets
//   K3 scatter_kernel : stable counting-sort scatter (wave shuffle ranks)
// ---------------------------------------------------------------------------

#define NTOK   8192
#define DIM    4096
#define NEXP   64
#define TOPK   8
#define TOK_PER_BLK 32                 // tokens per router block
#define NBLK   (NTOK / TOK_PER_BLK)    // 256 router/sort blocks
#define SLOTS_PER_BLK (TOK_PER_BLK * TOPK)  // 256 slots per sort block
#define KSPLIT (DIM / 2)               // K range per wave

typedef __attribute__((ext_vector_type(16))) __bf16 v16bf;
typedef __attribute__((ext_vector_type(8)))  float  v8f;

// Split 16 fp32 values (p0[0..7], p1[0..7]) into bf16 hi + bf16 residual lo.
// hi = rne_bf16(f); lo = rne_bf16(f - hi)  ->  3-term product ~ fp32 accuracy.
__device__ __forceinline__ void cvt_split16(const float* __restrict__ p0,
                                            const float* __restrict__ p1,
                                            v16bf& hi, v16bf& lo) {
#pragma unroll
  for (int i = 0; i < 8; ++i) {
    float f0 = p0[i], f1 = p1[i];
    __bf16 h0 = (__bf16)f0, h1 = (__bf16)f1;
    hi[i]     = h0;
    hi[i + 8] = h1;
    lo[i]     = (__bf16)(f0 - (float)h0);
    lo[i + 8] = (__bf16)(f1 - (float)h1);
  }
}

// ---------------------------------------------------------------------------
// Kernel 0: one-time gate_w split. 64*4096 elements, done once per launch;
// result (1 MB) stays resident in the 192 MB L2 for the whole router pass.
// ---------------------------------------------------------------------------
__global__ __launch_bounds__(256) void split_w_kernel(
    const float* __restrict__ gw,
    __bf16*      __restrict__ w_hi,
    __bf16*      __restrict__ w_lo)
{
  const int i = blockIdx.x * 256 + threadIdx.x;   // grid sized exactly
  float f = gw[i];
  __bf16 h = (__bf16)f;
  w_hi[i] = h;
  w_lo[i] = (__bf16)(f - (float)h);
}

// ---------------------------------------------------------------------------
// Kernel 1: GEMM + softmax + top-8 + per-block histogram.
// Block = 128 threads = 4 waves = 2 token-tiles x 2 K-halves.
// Each wave: 16 tokens x 64 experts (4 WMMA accumulators) over half of K.
// ---------------------------------------------------------------------------
__global__ __launch_bounds__(128) void router_kernel(
    const float*  __restrict__ x,        // [NTOK][DIM]
    const float*  __restrict__ bias,     // [NEXP]
    const __bf16* __restrict__ w_hi,     // [NEXP][DIM]
    const __bf16* __restrict__ w_lo,     // [NEXP][DIM]
    int*          __restrict__ sel,      // [NTOK*TOPK]
    float*        __restrict__ tsc,      // [NTOK*TOPK]
    int*          __restrict__ blockHist)// [NBLK][NEXP]
{
  __shared__ float lds_logits[TOK_PER_BLK][NEXP + 1];  // +1 pad: no bank conflicts
  __shared__ float bias_s[NEXP];
  __shared__ int   hist_s[NEXP];

  const int tid  = threadIdx.x;
  const int wave = tid >> 5;
  const int lane = tid & 31;
  const int hi16 = lane >> 4;      // half-wave (K-split selector in fragment)
  const int mrow = lane & 15;      // row of A / column of B
  const int tt   = wave & 1;       // token tile 0..1
  const int ks   = wave >> 1;      // K half 0..1
  const int tokenBase = blockIdx.x * TOK_PER_BLK + tt * 16;
  const int kBase = ks * KSPLIT;

  if (tid < NEXP) { bias_s[tid] = bias[tid]; hist_s[tid] = 0; }

  // A fragment (16x32 bf16): lane<16 holds K 0..15, lane>=16 K 16..31
  const float* xrow = x + (size_t)(tokenBase + mrow) * DIM + kBase + hi16 * 16;
  // B fragment (32x16 bf16): lane<16 holds K 0..15, lane>=16 K 16..31 (contiguous 32B)
  const __bf16* whrow = w_hi + (size_t)mrow * DIM + kBase + hi16 * 16;
  const __bf16* wlrow = w_lo + (size_t)mrow * DIM + kBase + hi16 * 16;

  v8f acc[4] = {};

  for (int kb = 0; kb < KSPLIT; kb += 32) {
    __builtin_prefetch(xrow + kb + 256, 0, 0);     // global_prefetch_b8
    v16bf ah, al;
    cvt_split16(xrow + kb, xrow + kb + 8, ah, al);  // x split once per wave
#pragma unroll
    for (int nt = 0; nt < 4; ++nt) {
      // nt*16*DIM*2B <= 384 KB: folds into the 24-bit instruction offset
      v16bf bh = *(const v16bf*)(whrow + nt * 16 * DIM + kb);
      v16bf bl = *(const v16bf*)(wlrow + nt * 16 * DIM + kb);
      // D = A_hi*B_hi + A_hi*B_lo + A_lo*B_hi + C   (drop O(2^-16) lo*lo)
      acc[nt] = __builtin_amdgcn_wmma_f32_16x16x32_bf16(false, ah, false, bh,
                                                        (short)0, acc[nt], false, false);
      acc[nt] = __builtin_amdgcn_wmma_f32_16x16x32_bf16(false, ah, false, bl,
                                                        (short)0, acc[nt], false, false);
      acc[nt] = __builtin_amdgcn_wmma_f32_16x16x32_bf16(false, al, false, bh,
                                                        (short)0, acc[nt], false, false);
    }
  }

  // C/D layout: VGPR r -> M = r (lanes 0-15) or r+8 (lanes 16-31); N = lane&15.
  // K-split reduction through LDS: ks=0 stores, ks=1 accumulates.
  if (ks == 0) {
#pragma unroll
    for (int nt = 0; nt < 4; ++nt)
#pragma unroll
      for (int r = 0; r < 8; ++r)
        lds_logits[tt * 16 + r + hi16 * 8][nt * 16 + mrow] = acc[nt][r];
  }
  __syncthreads();
  if (ks == 1) {
#pragma unroll
    for (int nt = 0; nt < 4; ++nt)
#pragma unroll
      for (int r = 0; r < 8; ++r)
        lds_logits[tt * 16 + r + hi16 * 8][nt * 16 + mrow] += acc[nt][r];
  }
  __syncthreads();

  // --- softmax + top-8 + histogram: one lane per token -------------------
  if (tid < TOK_PER_BLK) {
    float* srow = &lds_logits[tid][0];
    const int token = blockIdx.x * TOK_PER_BLK + tid;

    float m = -__builtin_inff();
#pragma unroll 4
    for (int e = 0; e < NEXP; ++e) m = fmaxf(m, srow[e]);

    float sum = 0.0f;
#pragma unroll 4
    for (int e = 0; e < NEXP; ++e) { float v = __expf(srow[e] - m); srow[e] = v; sum += v; }
    const float inv = 1.0f / sum;

    // 8 argmax passes; strict '>' scanning e=0..63 => ties pick lowest index,
    // matching jax.lax.top_k. Select on (score + bias), gather unbiased score.
    for (int k = 0; k < TOPK; ++k) {
      float best = -__builtin_inff();
      int   bi   = 0;
      for (int e = 0; e < NEXP; ++e) {
        float sc = srow[e] * inv + bias_s[e];
        if (sc > best) { best = sc; bi = e; }
      }
      float score = srow[bi] * inv;
      srow[bi] = -__builtin_inff();              // knock out for next pass
      sel[token * TOPK + k] = bi;
      tsc[token * TOPK + k] = score;
      atomicAdd(&hist_s[bi], 1);
    }
  }
  __syncthreads();

  if (tid < NEXP) blockHist[blockIdx.x * NEXP + tid] = hist_s[tid];
}

// ---------------------------------------------------------------------------
// Kernel 2: per-expert totals + exclusive scans -> per-block scatter offsets.
// ---------------------------------------------------------------------------
__global__ __launch_bounds__(64) void prefix_kernel(
    const int* __restrict__ blockHist,   // [NBLK][NEXP]
    int*       __restrict__ blockOffset, // [NBLK][NEXP]
    float*     __restrict__ out_counts)  // [NEXP]
{
  __shared__ int totals[NEXP];
  __shared__ int base[NEXP];
  const int e = threadIdx.x;

  int run = 0;
  for (int b = 0; b < NBLK; ++b) {
    int v = blockHist[b * NEXP + e];
    blockOffset[b * NEXP + e] = run;     // within-expert prefix over blocks
    run += v;
  }
  totals[e] = run;
  __syncthreads();

  if (e == 0) {                          // exclusive scan over experts
    int r = 0;
    for (int i = 0; i < NEXP; ++i) { base[i] = r; r += totals[i]; }
  }
  __syncthreads();

  const int be = base[e];
  for (int b = 0; b < NBLK; ++b) blockOffset[b * NEXP + e] += be;
  out_counts[e] = (float)totals[e];
}

// ---------------------------------------------------------------------------
// Kernel 3: stable counting-sort scatter. Block = 256 slots = 8 waves.
// rank(i) = blockOffset[b][e] + (#slots with e in earlier waves) + (#earlier
// lanes in my wave with e), the last via wave32 shuffle match-masks.
// ---------------------------------------------------------------------------
__global__ __launch_bounds__(256) void scatter_kernel(
    const int*   __restrict__ sel,
    const float* __restrict__ tsc,
    const int*   __restrict__ blockOffset,
    float*       __restrict__ out_scores,  // [NTOK*TOPK]
    float*       __restrict__ out_idx)     // [NTOK*TOPK] (token index as float)
{
  __shared__ int waveHist[8 * NEXP];
  const int t = threadIdx.x;
  const int b = blockIdx.x;

  waveHist[t] = 0; waveHist[t + 256] = 0;   // zero 512 entries with 256 threads
  __syncthreads();

  const int i    = b * SLOTS_PER_BLK + t;
  const int e    = sel[i];
  const int wave = t >> 5;
  const int lane = t & 31;

  unsigned mask = 0;
#pragma unroll
  for (int j = 0; j < 32; ++j) {
    int ev = __shfl(e, j, 32);
    mask |= (ev == e ? 1u : 0u) << j;
  }
  const int below = __popc(mask & ((1u << lane) - 1u));
  if (below == 0) waveHist[wave * NEXP + e] = __popc(mask);  // one writer per (wave,e)
  __syncthreads();

  if (t < NEXP) {                          // exclusive scan over waves, per expert
    int run = 0;
    for (int w = 0; w < 8; ++w) {
      int v = waveHist[w * NEXP + t];
      waveHist[w * NEXP + t] = run;
      run += v;
    }
  }
  __syncthreads();

  const int rank = blockOffset[b * NEXP + e] + waveHist[wave * NEXP + e] + below;
  out_scores[rank] = tsc[i];
  out_idx[rank]    = (float)(i >> 3);      // order[rank] // TOP_K
}

// ---------------------------------------------------------------------------
extern "C" void kernel_launch(void* const* d_in, const int* in_sizes, int n_in,
                              void* d_out, int out_size, void* d_ws, size_t ws_size,
                              hipStream_t stream) {
  const float* x    = (const float*)d_in[0];   // [8192*4096]
  const float* bias = (const float*)d_in[1];   // [64]
  const float* gw   = (const float*)d_in[2];   // [64*4096]

  // workspace layout (~1.63 MB)
  char*   ws   = (char*)d_ws;
  __bf16* w_hi = (__bf16*)ws;                               // 512 KB
  __bf16* w_lo = w_hi + (size_t)NEXP * DIM;                 // 512 KB
  int*    sel  = (int*)  (ws + (size_t)NEXP * DIM * 4);     // 256 KB
  float*  tsc  = (float*)(ws + (size_t)NEXP * DIM * 4 + (size_t)NTOK * TOPK * 4); // 256 KB
  int*    blockHist   = (int*)(ws + (size_t)NEXP * DIM * 4 + (size_t)NTOK * TOPK * 8);
  int*    blockOffset = blockHist + NBLK * NEXP;

  float* out        = (float*)d_out;
  float* out_scores = out;                       // [65536]
  float* out_idx    = out + NTOK * TOPK;         // [65536]
  float* out_counts = out + 2 * NTOK * TOPK;     // [64]

  split_w_kernel<<<(NEXP * DIM) / 256, 256, 0, stream>>>(gw, w_hi, w_lo);
  router_kernel <<<NBLK, 128, 0, stream>>>(x, bias, w_hi, w_lo, sel, tsc, blockHist);
  prefix_kernel <<<1,    64,  0, stream>>>(blockHist, blockOffset, out_counts);
  scatter_kernel<<<NBLK, 256, 0, stream>>>(sel, tsc, blockOffset, out_scores, out_idx);
}
